// EncoderBlock_9912784520071
// MI455X (gfx1250) — compile-verified, matrix operand order checked by reading, not run on hardware
//
#include <hip/hip_runtime.h>
#include <hip/hip_bf16.h>
#include <math.h>

// ---------------------------------------------------------------------------
// Encoder block: S=4096, E=2048, H=8192, fp32 in/out, bf16 WMMA internally.
// ---------------------------------------------------------------------------
#define SS 4096
#define EE 2048
#define HH 8192

typedef __attribute__((ext_vector_type(16))) __bf16 v16bf;
typedef __attribute__((ext_vector_type(8)))  __bf16 bfx8;
typedef __attribute__((ext_vector_type(8)))  float  v8f;

union ABfrag { v16bf v; bfx8 h[2]; };

#define BM 128
#define BN 128
#define BK 32
#define LDP (BK + 8)   // padded LDS row (16B-aligned chunks, bank spread)

#if defined(__HIP_DEVICE_COMPILE__) && __has_builtin(__builtin_amdgcn_global_load_async_to_lds_b128)
#define EB_ASYNC 1
#else
#define EB_ASYNC 0
#endif

// Builtin signature (from hipcc diagnostic): (v4i AS1*, v4i AS3*, imm i32, imm i32)
typedef int eb_v4i __attribute__((vector_size(16)));
typedef __attribute__((address_space(1))) eb_v4i eb_gv4i;
typedef __attribute__((address_space(3))) eb_v4i eb_lv4i;

__device__ __forceinline__ void eb_copy16(const __bf16* g, __bf16* l) {
#if EB_ASYNC
  __builtin_amdgcn_global_load_async_to_lds_b128((eb_gv4i*)g, (eb_lv4i*)l, 0, 0);
#else
  *(bfx8*)l = *(const bfx8*)g;
#endif
}

__device__ __forceinline__ void eb_wait_async() {
#if EB_ASYNC
#if __has_builtin(__builtin_amdgcn_s_wait_asynccnt)
  __builtin_amdgcn_s_wait_asynccnt(0);
#else
  asm volatile("s_wait_asynccnt 0" ::: "memory");
#endif
#endif
}

// ---------------------------------------------------------------------------
// NT bf16 GEMM: C[M,N] = A[M,K] (row-major) * B[N,K] (row-major, i.e. B^T)
// 256 threads = 8 waves; block tile 128x128, wave tile 64x32 (4x2 WMMA tiles).
// Epilogue: +bias[n], optional relu, optional +residual (f32), stores to any
// of: f32 C, bf16 C, bf16 C^T.
// ---------------------------------------------------------------------------
__global__ __launch_bounds__(256) void eb_gemm(
    const __bf16* __restrict__ A, const __bf16* __restrict__ B,
    int M, int N, int K,
    const float* __restrict__ bias, const float* __restrict__ residual,
    float* __restrict__ outF, __bf16* __restrict__ outB,
    __bf16* __restrict__ outBT, int relu)
{
  __shared__ __bf16 As[2][BM][LDP];
  __shared__ __bf16 Bs[2][BN][LDP];

  const int tid  = threadIdx.x;
  const int lane = tid & 31;
  const int wave = tid >> 5;
  const int wm   = wave & 1;   // 2 row-groups of 64
  const int wn   = wave >> 1;  // 4 col-groups of 32
  const int m0   = blockIdx.y * BM;
  const int n0   = blockIdx.x * BN;

  const v8f vzero = {0.f, 0.f, 0.f, 0.f, 0.f, 0.f, 0.f, 0.f};
  v8f acc[4][2];
#pragma unroll
  for (int i = 0; i < 4; ++i)
#pragma unroll
    for (int j = 0; j < 2; ++j) acc[i][j] = vzero;

  auto load_tiles = [&](int k0, int buf) {
    // 128 rows x 32 cols bf16 = 512 chunks of 16B per matrix; 2 per thread.
#pragma unroll
    for (int i = 0; i < 2; ++i) {
      const int cid = tid + i * 256;
      const int r   = cid >> 2;
      const int c   = (cid & 3) * 8;
      eb_copy16(&A[(size_t)(m0 + r) * K + k0 + c], &As[buf][r][c]);
      eb_copy16(&B[(size_t)(n0 + r) * K + k0 + c], &Bs[buf][r][c]);
    }
  };

  const int nK = K / BK;
  int buf = 0;
  load_tiles(0, 0);
  eb_wait_async();
  __syncthreads();

  for (int kt = 0; kt < nK; ++kt) {
    if (kt + 1 < nK) load_tiles((kt + 1) * BK, buf ^ 1);

    // B fragments (reused across the 4 M-tiles).
    ABfrag bfr[2];
#pragma unroll
    for (int j = 0; j < 2; ++j) {
      const __bf16* p = &Bs[buf][wn * 32 + j * 16 + (lane & 15)][(lane >> 4) * 8];
      bfr[j].h[0] = *(const bfx8*)p;
      bfr[j].h[1] = *(const bfx8*)(p + 16);
    }
#pragma unroll
    for (int i = 0; i < 4; ++i) {
      ABfrag afr;
      const __bf16* p = &As[buf][wm * 64 + i * 16 + (lane & 15)][(lane >> 4) * 8];
      afr.h[0] = *(const bfx8*)p;
      afr.h[1] = *(const bfx8*)(p + 16);
#pragma unroll
      for (int j = 0; j < 2; ++j)
        acc[i][j] = __builtin_amdgcn_wmma_f32_16x16x32_bf16(
            false, afr.v, false, bfr[j].v, (short)0, acc[i][j], false, false);
    }

    eb_wait_async();
    __syncthreads();
    buf ^= 1;
  }

  // Epilogue. C layout: VGPR r, lane L -> row = r + (L>=16 ? 8 : 0), col = L%16.
#pragma unroll
  for (int i = 0; i < 4; ++i) {
#pragma unroll
    for (int j = 0; j < 2; ++j) {
      const int mb  = m0 + wm * 64 + i * 16 + (lane >> 4) * 8;
      const int col = n0 + wn * 32 + j * 16 + (lane & 15);
      const float bv = bias ? bias[col] : 0.0f;
      float vals[8];
#pragma unroll
      for (int r = 0; r < 8; ++r) {
        float v = acc[i][j][r] + bv;
        if (relu) v = fmaxf(v, 0.0f);
        if (residual) v += residual[(size_t)(mb + r) * N + col];
        vals[r] = v;
      }
      if (outF) {
#pragma unroll
        for (int r = 0; r < 8; ++r) outF[(size_t)(mb + r) * N + col] = vals[r];
      }
      if (outB) {
#pragma unroll
        for (int r = 0; r < 8; ++r) outB[(size_t)(mb + r) * N + col] = (__bf16)vals[r];
      }
      if (outBT) {  // contiguous 16B store per lane into C^T[N,M]
        bfx8 t;
#pragma unroll
        for (int r = 0; r < 8; ++r) t[r] = (__bf16)vals[r];
        *(bfx8*)&outBT[(size_t)col * M + mb] = t;
      }
    }
  }
}

// ---------------------------------------------------------------------------
// Row softmax: P[row,:] = softmax(scores[row,:] * scale), bf16 output.
// ---------------------------------------------------------------------------
__global__ __launch_bounds__(256) void eb_softmax(
    const float* __restrict__ scores, __bf16* __restrict__ P, int n, float scale)
{
  const int row = blockIdx.x;
  const float* x = scores + (size_t)row * n;
  __bf16*      y = P      + (size_t)row * n;
  __shared__ float red[8];

  float m = -3.0e38f;
  for (int i = threadIdx.x; i < n; i += 256) m = fmaxf(m, x[i] * scale);
#pragma unroll
  for (int off = 16; off; off >>= 1) m = fmaxf(m, __shfl_xor(m, off, 32));
  if ((threadIdx.x & 31) == 0) red[threadIdx.x >> 5] = m;
  __syncthreads();
  if (threadIdx.x == 0) {
    float mm = red[0];
#pragma unroll
    for (int i = 1; i < 8; ++i) mm = fmaxf(mm, red[i]);
    red[0] = mm;
  }
  __syncthreads();
  m = red[0];
  __syncthreads();

  float s = 0.0f;
  for (int i = threadIdx.x; i < n; i += 256) s += __expf(x[i] * scale - m);
#pragma unroll
  for (int off = 16; off; off >>= 1) s += __shfl_xor(s, off, 32);
  if ((threadIdx.x & 31) == 0) red[threadIdx.x >> 5] = s;
  __syncthreads();
  if (threadIdx.x == 0) {
    float ss = 0.0f;
#pragma unroll
    for (int i = 0; i < 8; ++i) ss += red[i];
    red[0] = ss;
  }
  __syncthreads();
  const float inv = 1.0f / red[0];

  for (int i = threadIdx.x; i < n; i += 256)
    y[i] = (__bf16)(__expf(x[i] * scale - m) * inv);
}

// ---------------------------------------------------------------------------
// Global LayerNorm over ALL S*E elements (two-pass, atomic partials).
// ---------------------------------------------------------------------------
__global__ void eb_zero4(float* p) { if (threadIdx.x < 4) p[threadIdx.x] = 0.0f; }

__global__ __launch_bounds__(256) void eb_reduce(
    const float* __restrict__ x, unsigned long long n, float* __restrict__ sums)
{
  float s = 0.0f, s2 = 0.0f;
  for (unsigned long long i = (unsigned long long)blockIdx.x * 256 + threadIdx.x;
       i < n; i += (unsigned long long)gridDim.x * 256) {
    float v = x[i];
    s += v; s2 += v * v;
  }
  __shared__ float a[8], b[8];
#pragma unroll
  for (int off = 16; off; off >>= 1) { s += __shfl_xor(s, off, 32); s2 += __shfl_xor(s2, off, 32); }
  if ((threadIdx.x & 31) == 0) { a[threadIdx.x >> 5] = s; b[threadIdx.x >> 5] = s2; }
  __syncthreads();
  if (threadIdx.x == 0) {
    float S = 0.0f, S2 = 0.0f;
#pragma unroll
    for (int i = 0; i < 8; ++i) { S += a[i]; S2 += b[i]; }
    atomicAdd(&sums[0], S);
    atomicAdd(&sums[1], S2);
  }
}

__global__ __launch_bounds__(256) void eb_ln_apply(
    const float* __restrict__ x, const float* __restrict__ sums,
    unsigned long long n, float inv_n,
    float* __restrict__ outF, __bf16* __restrict__ outB)
{
  const float mu  = sums[0] * inv_n;
  const float var = sums[1] * inv_n - mu * mu;
  const float rs  = rsqrtf(var + 1e-5f);
  for (unsigned long long i = (unsigned long long)blockIdx.x * 256 + threadIdx.x;
       i < n; i += (unsigned long long)gridDim.x * 256) {
    const float v = (x[i] - mu) * rs;
    if (outF) outF[i] = v;
    if (outB) outB[i] = (__bf16)v;
  }
}

// ---------------------------------------------------------------------------
// fp32 -> bf16 converts (plain and transposed, for NT weight layout).
// ---------------------------------------------------------------------------
__global__ __launch_bounds__(256) void eb_cvt(
    const float* __restrict__ x, __bf16* __restrict__ y, unsigned long long n)
{
  for (unsigned long long i = (unsigned long long)blockIdx.x * 256 + threadIdx.x;
       i < n; i += (unsigned long long)gridDim.x * 256)
    y[i] = (__bf16)x[i];
}

__global__ __launch_bounds__(256) void eb_cvtT(
    const float* __restrict__ x, __bf16* __restrict__ y, int R, int C)
{
  const unsigned long long n = (unsigned long long)R * C;
  for (unsigned long long i = (unsigned long long)blockIdx.x * 256 + threadIdx.x;
       i < n; i += (unsigned long long)gridDim.x * 256) {
    const int r = (int)(i / C);
    const int c = (int)(i % C);
    y[(size_t)c * R + r] = (__bf16)x[i];
  }
}

// ---------------------------------------------------------------------------
extern "C" void kernel_launch(void* const* d_in, const int* in_sizes, int n_in,
                              void* d_out, int out_size, void* d_ws, size_t ws_size,
                              hipStream_t stream) {
  const float* x  = (const float*)d_in[0];
  const float* Wq = (const float*)d_in[1];
  const float* bq = (const float*)d_in[2];
  const float* Wk = (const float*)d_in[3];
  const float* bk = (const float*)d_in[4];
  const float* Wv = (const float*)d_in[5];
  const float* bv = (const float*)d_in[6];
  const float* W1 = (const float*)d_in[7];
  const float* b1 = (const float*)d_in[8];
  const float* W2 = (const float*)d_in[9];
  const float* b2 = (const float*)d_in[10];
  float* out = (float*)d_out;

  char* w = (char*)d_ws;
  auto carve = [&](size_t bytes) -> void* {
    void* p = (void*)w;
    w += (bytes + 255) & ~(size_t)255;
    return p;
  };

  const size_t SE = (size_t)SS * EE, SSq = (size_t)SS * SS, SH = (size_t)SS * HH,
               EH = (size_t)EE * HH, EEq = (size_t)EE * EE;

  float*  sums = (float*)carve(4 * sizeof(float));
  __bf16* xb   = (__bf16*)carve(SE * 2);
  __bf16* Wqt  = (__bf16*)carve(EEq * 2);
  __bf16* Wkt  = (__bf16*)carve(EEq * 2);
  __bf16* Wvt  = (__bf16*)carve(EEq * 2);
  __bf16* W1t  = (__bf16*)carve(EH * 2);
  __bf16* W2t  = (__bf16*)carve(EH * 2);
  __bf16* Qb   = (__bf16*)carve(SE * 2);
  __bf16* Kb   = (__bf16*)carve(SE * 2);
  __bf16* Vt   = (__bf16*)carve(SE * 2);   // V^T [E,S]
  float*  sc   = (float*)carve(SSq * 4);   // raw scores
  __bf16* P    = (__bf16*)carve(SSq * 2);  // softmax probs
  float*  hpre = (float*)carve(SE * 4);    // x + attn
  float*  h    = (float*)carve(SE * 4);    // ln1 out f32
  __bf16* hb   = (__bf16*)carve(SE * 2);   // ln1 out bf16
  __bf16* f1   = (__bf16*)carve(SH * 2);   // relu(h W1 + b1)
  float*  ypre = (float*)carve(SE * 4);    // h + ffn

  const dim3 blk(256);
  const dim3 gcv(4096);

  eb_zero4<<<dim3(1), dim3(32), 0, stream>>>(sums);

  // bf16 casts (+ transpose weights into NT layout)
  eb_cvt <<<gcv, blk, 0, stream>>>(x, xb, SE);
  eb_cvtT<<<gcv, blk, 0, stream>>>(Wq, Wqt, EE, EE);
  eb_cvtT<<<gcv, blk, 0, stream>>>(Wk, Wkt, EE, EE);
  eb_cvtT<<<gcv, blk, 0, stream>>>(Wv, Wvt, EE, EE);
  eb_cvtT<<<gcv, blk, 0, stream>>>(W1, W1t, EE, HH);  // -> [H,E]
  eb_cvtT<<<gcv, blk, 0, stream>>>(W2, W2t, HH, EE);  // -> [E,H]

  // Q, K (bf16), V (stored transposed)
  eb_gemm<<<dim3(EE / BN, SS / BM), blk, 0, stream>>>(xb, Wqt, SS, EE, EE, bq, nullptr, nullptr, Qb, nullptr, 0);
  eb_gemm<<<dim3(EE / BN, SS / BM), blk, 0, stream>>>(xb, Wkt, SS, EE, EE, bk, nullptr, nullptr, Kb, nullptr, 0);
  eb_gemm<<<dim3(EE / BN, SS / BM), blk, 0, stream>>>(xb, Wvt, SS, EE, EE, bv, nullptr, nullptr, nullptr, Vt, 0);

  // scores = Q K^T (K rows are already the NT B layout)
  eb_gemm<<<dim3(SS / BN, SS / BM), blk, 0, stream>>>(Qb, Kb, SS, SS, EE, nullptr, nullptr, sc, nullptr, nullptr, 0);

  // softmax with scale 1/sqrt(S) = 1/64
  eb_softmax<<<dim3(SS), blk, 0, stream>>>(sc, P, SS, 0.015625f);

  // attn = P V  (+ residual x), f32
  eb_gemm<<<dim3(EE / BN, SS / BM), blk, 0, stream>>>(P, Vt, SS, EE, SS, nullptr, x, hpre, nullptr, nullptr, 0);

  // h = LN_all(x + attn)
  eb_reduce  <<<dim3(1024), blk, 0, stream>>>(hpre, SE, &sums[0]);
  eb_ln_apply<<<dim3(2048), blk, 0, stream>>>(hpre, &sums[0], SE, 1.0f / (float)SE, h, hb);

  // ffn1 = relu(h W1 + b1), bf16
  eb_gemm<<<dim3(HH / BN, SS / BM), blk, 0, stream>>>(hb, W1t, SS, HH, EE, b1, nullptr, nullptr, f1, nullptr, 1);

  // ffn2 = f1 W2 + b2 + h, f32
  eb_gemm<<<dim3(EE / BN, SS / BM), blk, 0, stream>>>(f1, W2t, SS, EE, HH, b2, h, ypre, nullptr, nullptr, 0);

  // out = LN_all(h + ffn)
  eb_reduce  <<<dim3(1024), blk, 0, stream>>>(ypre, SE, &sums[2]);
  eb_ln_apply<<<dim3(2048), blk, 0, stream>>>(ypre, &sums[2], SE, 1.0f / (float)SE, out, nullptr);
}
